// SelfAttention_77163382440839
// MI455X (gfx1250) — matrix operand order repaired and checked by reading, not compiled, not run on hardware
//
#include <hip/hip_runtime.h>

// Problem sizes (fixed by the reference)
#define NB 8
#define NC 512
#define NL 2048
#define NK 64
#define MALL 640   // NK + NK + NC stacked projection rows

typedef __attribute__((ext_vector_type(16))) __bf16 v16bf;
typedef __attribute__((ext_vector_type(8)))  __bf16 v8bf;
typedef __attribute__((ext_vector_type(8)))  float  v8f;
typedef __attribute__((ext_vector_type(4)))  unsigned int v4u;
typedef __attribute__((ext_vector_type(8)))  int v8i;
typedef __attribute__((ext_vector_type(4)))  int v4i;

// ---------------------------------------------------------------------------
// WMMA fragment helpers (CDNA5 16x16x32 bf16 layouts, wave32)
//  A (16xK=32): lane l -> row l&15; elems e<8: K=8*hi+e ; e>=8: K=16+8*hi+(e-8)
//  B (K=32x16): lane l -> col l&15; elems e<8: K=8*hi+e ; e>=8: K=16+8*hi+(e-8)
//  C/D f32    : lane l -> col l&15; VGPR v -> row v+8*hi
// ---------------------------------------------------------------------------
__device__ __forceinline__ v16bf frag_a(const __bf16* rowbase, int hi) {
  const v8bf lo = *(const v8bf*)(rowbase + 8 * hi);
  const v8bf hh = *(const v8bf*)(rowbase + 16 + 8 * hi);
  v16bf r;
#pragma unroll
  for (int i = 0; i < 8; ++i) { r[i] = lo[i]; r[i + 8] = hh[i]; }
  return r;
}

__device__ __forceinline__ v16bf frag_b(const __bf16* colbase, int hi) {
  const v8bf lo = *(const v8bf*)(colbase + 8 * hi);
  const v8bf hh = *(const v8bf*)(colbase + 16 + 8 * hi);
  v16bf r;
#pragma unroll
  for (int i = 0; i < 8; ++i) { r[i] = lo[i]; r[i + 8] = hh[i]; }
  return r;
}

__device__ __forceinline__ v8f wmma_bf16(v16bf a, v16bf b, v8f c) {
  return __builtin_amdgcn_wmma_f32_16x16x32_bf16(
      /*neg_a=*/false, a, /*neg_b=*/false, b,
      /*c_mod=*/(short)0, c, /*reuse_a=*/false, /*reuse_b=*/false);
}

// ---------------------------------------------------------------------------
// Tensor Data Mover: 2D tile load (bf16) global -> LDS.
//  D# layout per CDNA5 ISA ch.8 (group0: count/lds_addr/global_addr/type=2;
//  group1: data_size=2B, tensor dims, tile dims, dim0 stride; groups 2/3 = 0).
// ---------------------------------------------------------------------------
#define HAVE_TDM __has_builtin(__builtin_amdgcn_tensor_load_to_lds)

#if HAVE_TDM
__device__ __forceinline__ void tdm_load_2d(unsigned lds_off, const __bf16* gptr,
                                            unsigned tensor_w, unsigned tensor_h,
                                            unsigned tile_w, unsigned tile_h,
                                            unsigned stride_elems) {
  const unsigned long long ga = (unsigned long long)(uintptr_t)gptr;
  v4u g0;
  g0[0] = 1u;                                              // count=1, user desc
  g0[1] = lds_off;                                         // lds_addr (bytes)
  g0[2] = (unsigned)(ga & 0xffffffffu);                    // global_addr[31:0]
  g0[3] = (unsigned)((ga >> 32) & 0x01ffffffu) | 0x80000000u; // [56:32] + type=2

  v8i g1;
  g1[0] = (int)(1u << 16);                                 // data_size=1 (2 B)
  g1[1] = (int)((tensor_w & 0xffffu) << 16);               // tensor_dim0[15:0]
  g1[2] = (int)(((tensor_w >> 16) & 0xffffu) |
                ((tensor_h & 0xffffu) << 16));             // dim0[31:16]|dim1 lo
  g1[3] = (int)(((tensor_h >> 16) & 0xffffu) |
                ((tile_w & 0xffffu) << 16));               // dim1 hi | tile_dim0
  g1[4] = (int)(tile_h & 0xffffu);                         // tile_dim1 (dim2=0)
  g1[5] = (int)stride_elems;                               // dim0_stride[31:0]
  g1[6] = 0;                                               // stride hi, dim1 st
  g1[7] = 0;

  v4i z4 = {0, 0, 0, 0};
#if defined(__clang_major__) && __clang_major__ >= 23
  v8i z8 = {0, 0, 0, 0, 0, 0, 0, 0};
  __builtin_amdgcn_tensor_load_to_lds(g0, g1, z4, z4, z8, 0);
#else
  __builtin_amdgcn_tensor_load_to_lds(g0, g1, z4, z4, 0);
#endif
}
#endif

// ---------------------------------------------------------------------------
// Kernel 0a: x (B,C,L) f32 -> xT (B,L,C) bf16   (K-contiguous for proj GEMM)
// ---------------------------------------------------------------------------
__global__ void __launch_bounds__(256)
k_cvt_x(const float* __restrict__ x, __bf16* __restrict__ xT) {
  int idx = blockIdx.x * blockDim.x + threadIdx.x;
  if (idx >= NB * NC * NL) return;
  int b = idx / (NC * NL);
  int r = idx % (NC * NL);
  int c = r / NL;
  int l = r % NL;
  xT[(size_t)b * NL * NC + (size_t)l * NC + c] = (__bf16)x[idx];
}

// ---------------------------------------------------------------------------
// Kernel 0b: stack [Wf;Wg;Wh] -> Wb (640x512) bf16
// ---------------------------------------------------------------------------
__global__ void __launch_bounds__(256)
k_cvt_w(const float* __restrict__ Wf, const float* __restrict__ Wg,
        const float* __restrict__ Wh, __bf16* __restrict__ Wb) {
  int idx = blockIdx.x * blockDim.x + threadIdx.x;
  if (idx >= MALL * NC) return;
  int m = idx / NC, k = idx % NC;
  float v;
  if (m < NK)            v = Wf[m * NC + k];
  else if (m < 2 * NK)   v = Wg[(m - NK) * NC + k];
  else                   v = Wh[(m - 2 * NK) * NC + k];
  Wb[idx] = (__bf16)v;
}

// ---------------------------------------------------------------------------
// Kernel 1: projections  P = Wb (640x512) * x (512xL) + bias, per batch.
//  f,g written TRANSPOSED (L x 64) bf16; h written (C x L) bf16.
//  Block: 128 thr (4 waves); each wave: 16(M) x 64(N) tile; grid (L/64,10,B).
// ---------------------------------------------------------------------------
__global__ void __launch_bounds__(128)
k_proj(const __bf16* __restrict__ Wb, const __bf16* __restrict__ xT,
       const float* __restrict__ bf, const float* __restrict__ bg,
       const float* __restrict__ bh,
       __bf16* __restrict__ fT, __bf16* __restrict__ gT,
       __bf16* __restrict__ hB) {
  const int b   = blockIdx.z;
  const int by  = blockIdx.y;
  const int n0  = blockIdx.x * 64;
  const int w   = threadIdx.x >> 5;
  const int l   = threadIdx.x & 31;
  const int col = l & 15;
  const int hi  = l >> 4;
  const int mrow = by * 64 + w * 16 + col;   // A-matrix row this lane loads

  const __bf16* xb = xT + (size_t)b * NL * NC;

  v8f acc[4];
#pragma unroll
  for (int t = 0; t < 4; ++t)
#pragma unroll
    for (int e = 0; e < 8; ++e) acc[t][e] = 0.f;

  for (int kk = 0; kk < NC; kk += 32) {
    v16bf a = frag_a(Wb + (size_t)mrow * NC + kk, hi);
#pragma unroll
    for (int t = 0; t < 4; ++t) {
      const int n = n0 + t * 16 + col;
      v16bf bm = frag_b(xb + (size_t)n * NC + kk, hi);
      acc[t] = wmma_bf16(a, bm, acc[t]);
    }
  }

  // Epilogue: region is uniform per block (by==0 -> f, by==1 -> g, else h)
  if (by == 0) {
#pragma unroll
    for (int t = 0; t < 4; ++t) {
      const int n = n0 + t * 16 + col;
#pragma unroll
      for (int v = 0; v < 8; ++v) {
        const int m = w * 16 + v + 8 * hi;            // 0..63
        fT[(size_t)b * NL * NK + (size_t)n * NK + m] = (__bf16)(acc[t][v] + bf[m]);
      }
    }
  } else if (by == 1) {
#pragma unroll
    for (int t = 0; t < 4; ++t) {
      const int n = n0 + t * 16 + col;
#pragma unroll
      for (int v = 0; v < 8; ++v) {
        const int m = w * 16 + v + 8 * hi;            // 0..63
        gT[(size_t)b * NL * NK + (size_t)n * NK + m] = (__bf16)(acc[t][v] + bg[m]);
      }
    }
  } else {
    const int mh0 = by * 64 - 128;
#pragma unroll
    for (int t = 0; t < 4; ++t) {
      const int n = n0 + t * 16 + col;
#pragma unroll
      for (int v = 0; v < 8; ++v) {
        const int m = mh0 + w * 16 + v + 8 * hi;      // 0..511
        hB[(size_t)b * NC * NL + (size_t)m * NL + n] = (__bf16)(acc[t][v] + bh[m]);
      }
    }
  }
}

// ---------------------------------------------------------------------------
// Kernel 2: att = softmax_rows( f^T g ), stored bf16 (B,L,L).
//  Block: 256 thr (8 waves) per (batch, 16-query block). Each wave owns 256
//  columns = 16 WMMA tiles held in VGPRs; K=64 = two k=32 WMMA steps.
//  Softmax: shfl_xor within 16-lane halves + LDS cross-wave reduction.
// ---------------------------------------------------------------------------
__global__ void __launch_bounds__(256)
k_att(const __bf16* __restrict__ fT, const __bf16* __restrict__ gT,
      __bf16* __restrict__ att) {
  const int b   = blockIdx.y;
  const int qi  = blockIdx.x;        // 16-row query block
  const int w   = threadIdx.x >> 5;
  const int l   = threadIdx.x & 31;
  const int col = l & 15;
  const int hi  = l >> 4;

  const __bf16* fb = fT + (size_t)b * NL * NK;
  const __bf16* gb = gT + (size_t)b * NL * NK;

  const v16bf a0 = frag_a(fb + (size_t)(qi * 16 + col) * NK + 0,  hi);
  const v16bf a1 = frag_a(fb + (size_t)(qi * 16 + col) * NK + 32, hi);

  v8f s[16];
#pragma unroll
  for (int t = 0; t < 16; ++t)
#pragma unroll
    for (int e = 0; e < 8; ++e) s[t][e] = 0.f;

#pragma unroll
  for (int t = 0; t < 16; ++t) {
    const int j = w * 256 + t * 16 + col;
    v16bf b0 = frag_b(gb + (size_t)j * NK + 0,  hi);
    v16bf b1 = frag_b(gb + (size_t)j * NK + 32, hi);
    s[t] = wmma_bf16(a0, b0, s[t]);
    s[t] = wmma_bf16(a1, b1, s[t]);
  }

  __shared__ float red[8][16];
  __shared__ float grow[16];

  // ---- row max ----
  float lmax[8];
#pragma unroll
  for (int v = 0; v < 8; ++v) {
    float m = s[0][v];
#pragma unroll
    for (int t = 1; t < 16; ++t) m = fmaxf(m, s[t][v]);
#pragma unroll
    for (int d = 1; d < 16; d <<= 1) m = fmaxf(m, __shfl_xor(m, d, 32));
    lmax[v] = m;
  }
  if (col == 0) {
#pragma unroll
    for (int v = 0; v < 8; ++v) red[w][8 * hi + v] = lmax[v];
  }
  __syncthreads();
  if (threadIdx.x < 16) {
    float m = red[0][threadIdx.x];
#pragma unroll
    for (int ww = 1; ww < 8; ++ww) m = fmaxf(m, red[ww][threadIdx.x]);
    grow[threadIdx.x] = m;
  }
  __syncthreads();
  float rmax[8];
#pragma unroll
  for (int v = 0; v < 8; ++v) rmax[v] = grow[8 * hi + v];

  // ---- exp + row sum ----
  float lsum[8];
#pragma unroll
  for (int v = 0; v < 8; ++v) lsum[v] = 0.f;
#pragma unroll
  for (int t = 0; t < 16; ++t) {
#pragma unroll
    for (int v = 0; v < 8; ++v) {
      float e = __expf(s[t][v] - rmax[v]);
      s[t][v] = e;
      lsum[v] += e;
    }
  }
#pragma unroll
  for (int v = 0; v < 8; ++v) {
    float m = lsum[v];
#pragma unroll
    for (int d = 1; d < 16; d <<= 1) m += __shfl_xor(m, d, 32);
    lsum[v] = m;
  }
  if (col == 0) {
#pragma unroll
    for (int v = 0; v < 8; ++v) red[w][8 * hi + v] = lsum[v];
  }
  __syncthreads();
  if (threadIdx.x < 16) {
    float m = 0.f;
#pragma unroll
    for (int ww = 0; ww < 8; ++ww) m += red[ww][threadIdx.x];
    grow[threadIdx.x] = m;
  }
  __syncthreads();
  float rinv[8];
#pragma unroll
  for (int v = 0; v < 8; ++v) rinv[v] = 1.f / grow[8 * hi + v];

  // ---- normalize + store bf16 ----
  __bf16* ab = att + (size_t)b * NL * NL;
#pragma unroll
  for (int t = 0; t < 16; ++t) {
    const int j = w * 256 + t * 16 + col;
#pragma unroll
    for (int v = 0; v < 8; ++v) {
      const int i = qi * 16 + v + 8 * hi;
      ab[(size_t)i * NL + j] = (__bf16)(s[t][v] * rinv[v]);
    }
  }
}

// ---------------------------------------------------------------------------
// Kernel 3: out[c,i] = sum_j h[c,j]*att[i,j] + x[c,i]   (K = L = 2048)
//  TDM version: wave 0 DMAs the 64x64 bf16 h-tile and att-tile into LDS
//  (double-buffered, TENSORcnt-tracked); all 4 waves consume WMMA fragments
//  from LDS. Block: 128 thr (4 waves); grid (L/64, C/64, B).
// ---------------------------------------------------------------------------
__global__ void __launch_bounds__(128)
k_out(const __bf16* __restrict__ hB, const __bf16* __restrict__ att,
      const float* __restrict__ x, float* __restrict__ out) {
  const int b   = blockIdx.z;
  const int cb  = blockIdx.y;
  const int n0  = blockIdx.x * 64;
  const int w   = threadIdx.x >> 5;
  const int l   = threadIdx.x & 31;
  const int col = l & 15;
  const int hi  = l >> 4;
  const int c0  = cb * 64;

  const __bf16* hb = hB + (size_t)b * NC * NL;
  const __bf16* ab = att + (size_t)b * NL * NL;

  v8f acc[4];
#pragma unroll
  for (int t = 0; t < 4; ++t)
#pragma unroll
    for (int e = 0; e < 8; ++e) acc[t][e] = 0.f;

#if HAVE_TDM
  __shared__ __bf16 hs[2][64 * 64];   // 8 KB per buffer
  __shared__ __bf16 as[2][64 * 64];   // 8 KB per buffer

  const unsigned hs_off[2] = { (unsigned)(uintptr_t)&hs[0][0],
                               (unsigned)(uintptr_t)&hs[1][0] };
  const unsigned as_off[2] = { (unsigned)(uintptr_t)&as[0][0],
                               (unsigned)(uintptr_t)&as[1][0] };

  const int NSTEPS = NL / 64;         // 32 K-steps of 64
  if (w == 0) {
    // prologue: stage step 0 into buffer 0
    tdm_load_2d(hs_off[0], hb + (size_t)c0 * NL + 0, NL, NC, 64, 64, NL);
    tdm_load_2d(as_off[0], ab + (size_t)n0 * NL + 0, NL, NL, 64, 64, NL);
  }

  for (int s = 0; s < NSTEPS; ++s) {
    const int buf = s & 1;
    if (w == 0) __builtin_amdgcn_s_wait_tensorcnt(0);
    __syncthreads();                              // buffer `buf` is ready
    if (w == 0 && (s + 1) < NSTEPS) {
      const int jj = (s + 1) * 64;
      tdm_load_2d(hs_off[buf ^ 1], hb + (size_t)c0 * NL + jj, NL, NC, 64, 64, NL);
      tdm_load_2d(as_off[buf ^ 1], ab + (size_t)n0 * NL + jj, NL, NL, 64, 64, NL);
    }
    const __bf16* ht = &hs[buf][0];
    const __bf16* at = &as[buf][0];
#pragma unroll
    for (int ks = 0; ks < 2; ++ks) {
      v16bf a = frag_a(ht + (w * 16 + col) * 64 + ks * 32, hi);
#pragma unroll
      for (int t = 0; t < 4; ++t) {
        v16bf bm = frag_b(at + (t * 16 + col) * 64 + ks * 32, hi);
        acc[t] = wmma_bf16(a, bm, acc[t]);
      }
    }
    __syncthreads();                              // all waves done with `buf`
  }
#else
  // Fallback: direct global-load pipeline (no TDM builtin on this toolchain)
  const __bf16* arow = hb + (size_t)(c0 + w * 16 + col) * NL;
  for (int jj = 0; jj < NL; jj += 32) {
    v16bf a = frag_a(arow + jj, hi);
    __builtin_prefetch((const void*)(arow + jj + 64), 0, 3);
#pragma unroll
    for (int t = 0; t < 4; ++t) {
      const int i = n0 + t * 16 + col;
      v16bf bm = frag_b(ab + (size_t)i * NL + jj, hi);
      acc[t] = wmma_bf16(a, bm, acc[t]);
    }
  }
#endif

  const float* xb = x + (size_t)b * NC * NL;
  float* ob = out + (size_t)b * NC * NL;
#pragma unroll
  for (int t = 0; t < 4; ++t) {
    const int i = n0 + t * 16 + col;
#pragma unroll
    for (int v = 0; v < 8; ++v) {
      const int c = c0 + w * 16 + v + 8 * hi;
      ob[(size_t)c * NL + i] = acc[t][v] + xb[(size_t)c * NL + i];
    }
  }
}

// ---------------------------------------------------------------------------
// Host launcher
// ---------------------------------------------------------------------------
extern "C" void kernel_launch(void* const* d_in, const int* in_sizes, int n_in,
                              void* d_out, int out_size, void* d_ws, size_t ws_size,
                              hipStream_t stream) {
  (void)in_sizes; (void)n_in; (void)out_size; (void)ws_size;

  const float* x  = (const float*)d_in[0];
  const float* Wf = (const float*)d_in[1];
  const float* bf = (const float*)d_in[2];
  const float* Wg = (const float*)d_in[3];
  const float* bg = (const float*)d_in[4];
  const float* Wh = (const float*)d_in[5];
  const float* bh = (const float*)d_in[6];
  float* out = (float*)d_out;

  char* ws = (char*)d_ws;
  size_t off = 0;
  __bf16* xT = (__bf16*)(ws + off); off += (size_t)NB * NL * NC * 2;   // 16 MB
  __bf16* Wb = (__bf16*)(ws + off); off += (size_t)MALL * NC * 2;      // 0.6 MB
  __bf16* fT = (__bf16*)(ws + off); off += (size_t)NB * NL * NK * 2;   // 2 MB
  __bf16* gT = (__bf16*)(ws + off); off += (size_t)NB * NL * NK * 2;   // 2 MB
  __bf16* hB = (__bf16*)(ws + off); off += (size_t)NB * NC * NL * 2;   // 16 MB
  __bf16* at = (__bf16*)(ws + off); off += (size_t)NB * NL * NL * 2;   // 64 MB

  // 0) precision conversion + layout transposes
  {
    int n = NB * NC * NL;
    k_cvt_x<<<(n + 255) / 256, 256, 0, stream>>>(x, xT);
  }
  {
    int n = MALL * NC;
    k_cvt_w<<<(n + 255) / 256, 256, 0, stream>>>(Wf, Wg, Wh, Wb);
  }

  // 1) projections f,g (transposed) and h
  k_proj<<<dim3(NL / 64, MALL / 64, NB), 128, 0, stream>>>(Wb, xT, bf, bg, bh,
                                                           fT, gT, hB);

  // 2) attention logits + softmax -> bf16 attention matrix
  k_att<<<dim3(NL / 16, NB), 256, 0, stream>>>(fT, gT, at);

  // 3) output GEMM + residual (TDM double-buffered LDS pipeline)
  k_out<<<dim3(NL / 64, NC / 64, NB), 128, 0, stream>>>(hB, at, x, out);
}
